// Encoder_56788057588483
// MI455X (gfx1250) — compile-verified, matrix operand order checked by reading, not run on hardware
//
#include <hip/hip_runtime.h>
#include <hip/hip_bf16.h>

// ---------------------------------------------------------------------------
// Bidirectional 3-layer LSTM, T=32768, F=256, H=128 (gates=512).
//   Setup : one-shot f32 -> (hi,lo) bf16 split of x and all W_ih
//   Phase A (per layer): xg[d] = X @ W_ih[d]^T + (b_ih+b_hh)  -> 3x bf16 WMMA
//   Phase B (per layer): sequential scan per direction; W_hh in LDS,
//                        xg rows double-buffered via GLOBAL_LOAD_ASYNC_TO_LDS;
//                        emits f32 h plus pre-split bf16 hi/lo for next layer
//   Final : emit layer-2 output at t = T-1 (256 floats)
// ---------------------------------------------------------------------------

#define SEQ_LEN   32768
#define NFEAT     256      // input width of every layer (F0=256, 2H=256)
#define HID       128
#define GATES     512      // 4*HID
#define W_STRIDE  132      // LDS row stride for W_hh: 2-way max bank conflict

typedef __attribute__((ext_vector_type(16))) __bf16 v16bf;
typedef __attribute__((ext_vector_type(8)))  __bf16 v8bf;
typedef __attribute__((ext_vector_type(4)))  __bf16 v4bf;
typedef __attribute__((ext_vector_type(8)))  float  v8f;

// ---------------------------------------------------------------------------
// One-shot f32 -> bf16 hi/lo split (used for x and for all W_ih at once).
// ---------------------------------------------------------------------------
__global__ __launch_bounds__(256)
void cvt_split(const float* __restrict__ src, __bf16* __restrict__ hi,
               __bf16* __restrict__ lo, int n4) {          // n4 = n/4
    int i = blockIdx.x * blockDim.x + threadIdx.x;
    if (i >= n4) return;
    float4 v = ((const float4*)src)[i];
    float f[4] = {v.x, v.y, v.z, v.w};
    v4bf h, l;
#pragma unroll
    for (int j = 0; j < 4; ++j) {
        __bf16 hb = (__bf16)f[j];
        h[j] = hb;
        l[j] = (__bf16)(f[j] - (float)hb);
    }
    ((v4bf*)hi)[i] = h;
    ((v4bf*)lo)[i] = l;
}

// ---------------------------------------------------------------------------
// Phase A: WMMA GEMM on pre-split bf16.  grid=(SEQ/64, GATES/64, 2), block=128.
// Each wave: 16(M) x 64(N) strip, K=256 in 8 blocks, 3 WMMAs per K-block
// (hi*hi + hi*lo + lo*hi restores ~fp32 accuracy).
// Fragment layout (ISA 16-bit A/B): lane&15 = row; lanes>=16 carry K+8 / K+24.
// ---------------------------------------------------------------------------
__device__ __forceinline__ v16bf load_frag(const __bf16* __restrict__ src,
                                           int row, int koff) {
    const __bf16* p = src + row * NFEAT + koff;
    v8bf a = *(const v8bf*)(p);          // K 0..7   (of this lane's half)
    v8bf b = *(const v8bf*)(p + 16);     // K 16..23
    return __builtin_shufflevector(a, b, 0,1,2,3,4,5,6,7,8,9,10,11,12,13,14,15);
}

__global__ __launch_bounds__(128)
void lstm_xproj_wmma(const __bf16* __restrict__ XH, const __bf16* __restrict__ XL,
                     const __bf16* __restrict__ WH, const __bf16* __restrict__ WL,
                     const float* __restrict__ bih, const float* __restrict__ bhh,
                     float* __restrict__ xg) {             // (2, SEQ, 512)
    const int dir  = blockIdx.z;
    const int lane = threadIdx.x & 31;
    const int wave = threadIdx.x >> 5;
    const int m0   = blockIdx.x * 64 + wave * 16;          // time rows
    const int n0   = blockIdx.y * 64;                      // gate cols

    const __bf16* WHd = WH + dir * (GATES * NFEAT);
    const __bf16* WLd = WL + dir * (GATES * NFEAT);
    const float*  bi  = bih + dir * GATES;
    const float*  bh  = bhh + dir * GATES;
    float*        Od  = xg  + (size_t)dir * SEQ_LEN * GATES;

    v8f acc[4];
#pragma unroll
    for (int nt = 0; nt < 4; ++nt)
#pragma unroll
        for (int j = 0; j < 8; ++j) acc[nt][j] = 0.0f;

    const int arow = m0 + (lane & 15);
    const int ksel = (lane >> 4) << 3;                     // +0 or +8

    for (int kb = 0; kb < 8; ++kb) {                       // K = 256
        const int koff = kb * 32 + ksel;
        v16bf ahi = load_frag(XH, arow, koff);
        v16bf alo = load_frag(XL, arow, koff);
#pragma unroll
        for (int nt = 0; nt < 4; ++nt) {
            const int brow = n0 + nt * 16 + (lane & 15);   // W_ih is N x K row-major
            v16bf bhi = load_frag(WHd, brow, koff);
            v16bf blo = load_frag(WLd, brow, koff);
            acc[nt] = __builtin_amdgcn_wmma_f32_16x16x32_bf16(
                false, ahi, false, bhi, (short)0, acc[nt], false, false);
            acc[nt] = __builtin_amdgcn_wmma_f32_16x16x32_bf16(
                false, ahi, false, blo, (short)0, acc[nt], false, false);
            acc[nt] = __builtin_amdgcn_wmma_f32_16x16x32_bf16(
                false, alo, false, bhi, (short)0, acc[nt], false, false);
        }
    }

    const int mbase = m0 + ((lane >> 4) << 3);             // C/D: elem j -> row
#pragma unroll
    for (int nt = 0; nt < 4; ++nt) {
        const int col  = n0 + nt * 16 + (lane & 15);
        const float bias = bi[col] + bh[col];
#pragma unroll
        for (int j = 0; j < 8; ++j)
            Od[(size_t)(mbase + j) * GATES + col] = acc[nt][j] + bias;
    }
}

// ---------------------------------------------------------------------------
// Async copy: one DWORD global -> LDS (GVS mode), tracked with ASYNCcnt.
// ---------------------------------------------------------------------------
__device__ __forceinline__ void async_g2lds_b32(unsigned lds_addr, unsigned voff,
                                                const float* base) {
    asm volatile("global_load_async_to_lds_b32 %0, %1, %2"
                 :: "v"(lds_addr), "v"(voff),
                    "s"((unsigned long long)(uintptr_t)base)
                 : "memory");
}
__device__ __forceinline__ void wait_async0() {
    asm volatile("s_wait_asynccnt 0" ::: "memory");
}

// ---------------------------------------------------------------------------
// Phase B: sequential scan.  grid = 2 (dir), block = 512 (16 waves, 1 WGP).
// LDS: W_hh 512x132 f32 | gates 512 | h 128 | xg double buffer 2x512  ~277 KB.
// Thread g owns gate g; threads <128 own (c,h).  Next timestep's xg row is
// fetched with async-to-LDS while the 512x128 matvec runs.
// ---------------------------------------------------------------------------
__device__ __forceinline__ float sigm(float x) { return 1.0f / (1.0f + __expf(-x)); }

__global__ __launch_bounds__(512)
void lstm_scan(const float* __restrict__ xg,     // (2, SEQ, 512), bias folded in
               const float* __restrict__ Whh,    // (2, 512, 128) for this layer
               float* __restrict__ out,          // (SEQ, 256) f32 (for final tap)
               __bf16* __restrict__ XH,          // (SEQ, 256) next-layer input hi
               __bf16* __restrict__ XL) {        // (SEQ, 256) next-layer input lo
    extern __shared__ float smem[];
    float* Wl    = smem;                          // 512 * 132
    float* g_lds = smem + GATES * W_STRIDE;       // 512
    float* h_lds = g_lds + GATES;                 // 128
    float* xbuf  = h_lds + HID;                   // 2 * 512 (double buffer)

    const int dir = blockIdx.x;
    const int tid = threadIdx.x;                  // 0..511 == gate index
    const float* Wd  = Whh + dir * (GATES * HID);
    const float* xgd = xg  + (size_t)dir * SEQ_LEN * GATES;

    // Stage W_hh into LDS (one padded row per thread, float4 moves).
#pragma unroll
    for (int kv = 0; kv < HID / 4; ++kv) {
        float4 w = *(const float4*)(Wd + tid * HID + kv * 4);
        *(float4*)&Wl[tid * W_STRIDE + kv * 4] = w;
    }
    if (tid < HID) h_lds[tid] = 0.0f;

    const unsigned xbuf_lds = (unsigned)(uintptr_t)xbuf + (unsigned)tid * 4u;
    // Prologue: fill buffer 0 with the first timestep's row.
    {
        const int t0 = (dir == 0) ? 0 : SEQ_LEN - 1;
        async_g2lds_b32(xbuf_lds, (unsigned)(((size_t)t0 * GATES + tid) * 4), xgd);
        wait_async0();
    }
    __syncthreads();

    float c_state = 0.0f;                         // live only in tid < 128
    const float* Wrow = &Wl[tid * W_STRIDE];

    for (int step = 0; step < SEQ_LEN; ++step) {
        const int t   = (dir == 0) ? step : (SEQ_LEN - 1 - step);
        const int cur = step & 1;

        // Kick off async fetch of the next row into the other buffer.
        const int tn = (dir == 0) ? (step + 1) : (SEQ_LEN - 2 - step);
        if (tn >= 0 && tn < SEQ_LEN)
            async_g2lds_b32(xbuf_lds + (unsigned)((cur ^ 1) * GATES * 4),
                            (unsigned)(((size_t)tn * GATES + tid) * 4), xgd);

        float acc = xbuf[cur * GATES + tid];
#pragma unroll
        for (int k = 0; k < HID; k += 4) {        // h broadcast: conflict-free
            float4 hv = *(const float4*)&h_lds[k];
            float4 wv = *(const float4*)&Wrow[k];
            acc = fmaf(hv.x, wv.x, acc);
            acc = fmaf(hv.y, wv.y, acc);
            acc = fmaf(hv.z, wv.z, acc);
            acc = fmaf(hv.w, wv.w, acc);
        }
        g_lds[tid] = acc;
        __syncthreads();

        if (tid < HID) {                          // torch gate order i, f, g, o
            float ig = sigm(g_lds[tid]);
            float fg = sigm(g_lds[HID + tid]);
            float gg = tanhf(g_lds[2 * HID + tid]);
            float og = sigm(g_lds[3 * HID + tid]);
            c_state  = fg * c_state + ig * gg;
            float h  = og * tanhf(c_state);
            h_lds[tid] = h;
            const size_t o = (size_t)t * 256 + dir * HID + tid;
            out[o] = h;
            __bf16 hb = (__bf16)h;                // pre-split for next layer GEMM
            XH[o] = hb;
            XL[o] = (__bf16)(h - (float)hb);
        }
        wait_async0();                            // own slot of next buffer ready
        __syncthreads();
    }
}

// ---------------------------------------------------------------------------
__global__ void lstm_final(const float* __restrict__ lastbuf, float* __restrict__ out) {
    int i = threadIdx.x;
    out[i] = lastbuf[(size_t)(SEQ_LEN - 1) * 256 + i];
}

// ---------------------------------------------------------------------------
extern "C" void kernel_launch(void* const* d_in, const int* in_sizes, int n_in,
                              void* d_out, int out_size, void* d_ws, size_t ws_size,
                              hipStream_t stream) {
    const float* x    = (const float*)d_in[0];   // (1, SEQ, 256)
    const float* W_ih = (const float*)d_in[1];   // (3, 2, 512, 256)
    const float* W_hh = (const float*)d_in[2];   // (3, 2, 512, 128)
    const float* b_ih = (const float*)d_in[3];   // (3, 2, 512)
    const float* b_hh = (const float*)d_in[4];   // (3, 2, 512)
    float*       out  = (float*)d_out;           // (1, 256)

    // Workspace: xg (2*SEQ*512 f32) | bufF (SEQ*256 f32) | XH | XL | WH | WL
    float*  xg   = (float*)d_ws;
    float*  bufF = xg + (size_t)2 * SEQ_LEN * GATES;
    __bf16* XH   = (__bf16*)(bufF + (size_t)SEQ_LEN * 256);
    __bf16* XL   = XH + (size_t)SEQ_LEN * NFEAT;
    __bf16* WH   = XL + (size_t)SEQ_LEN * NFEAT;
    __bf16* WL   = WH + (size_t)3 * 2 * GATES * NFEAT;

    // One-shot splits: layer-0 activations and all W_ih weights.
    {
        int n4 = SEQ_LEN * NFEAT / 4;
        cvt_split<<<(n4 + 255) / 256, 256, 0, stream>>>(x, XH, XL, n4);
        int w4 = 3 * 2 * GATES * NFEAT / 4;
        cvt_split<<<(w4 + 255) / 256, 256, 0, stream>>>(W_ih, WH, WL, w4);
    }

    const dim3 ggrid(SEQ_LEN / 64, GATES / 64, 2);
    const size_t smem_bytes =
        (size_t)(GATES * W_STRIDE + GATES + HID + 2 * GATES) * sizeof(float);

    for (int l = 0; l < 3; ++l) {
        lstm_xproj_wmma<<<ggrid, 128, 0, stream>>>(
            XH, XL,
            WH + (size_t)l * 2 * GATES * NFEAT,
            WL + (size_t)l * 2 * GATES * NFEAT,
            b_ih + (size_t)l * 2 * GATES,
            b_hh + (size_t)l * 2 * GATES,
            xg);
        // Scan overwrites XH/XL with this layer's output (next layer's input).
        lstm_scan<<<2, 512, smem_bytes, stream>>>(
            xg,
            W_hh + (size_t)l * 2 * GATES * HID,
            bufF, XH, XL);
    }
    lstm_final<<<1, 256, 0, stream>>>(bufF, out);
}
